// Decoder_with_attention_20005957665554
// MI455X (gfx1250) — matrix operand order repaired and checked by reading, not compiled, not run on hardware
//
#include <hip/hip_runtime.h>
#include <hip/hip_bf16.h>
#include <math.h>

// ---------------------------------------------------------------- constants
#define BATCH   64
#define PIX     196            // 14*14
#define ENC     2048
#define DEC     512
#define ATT     512
#define VOCAB   10000
#define TCAP    25
#define TSTEPS  24             // TCAP-1
#define XDIM    2560           // EMB + ENC

// GEMM tiling: block tile 64x64, 8 waves, each wave 16 rows x 32 cols (2 WMMAs)
#define BM 64
#define BN 64
#define BK 32
#define LDS_STRIDE 36          // BK + 4 pad: byte stride 72 -> conflict-free frag reads

typedef __attribute__((ext_vector_type(16))) __bf16 v16bf;
typedef __attribute__((ext_vector_type(8)))  float  v8f;

__device__ __forceinline__ float sigmoidf_(float x) {
    return 1.0f / (1.0f + __expf(-x));
}

// ---------------------------------------------------------------- staging helpers
// A tile: BM x BK = 2048 floats = 512 float4; 256 threads -> 2 quads each.
__device__ __forceinline__ void stage_load_A(const float* __restrict__ A, int lda,
                                             int m0, int k0, int K, int tid, float4 r[2])
{
    #pragma unroll
    for (int i = 0; i < 2; ++i) {
        int idx = tid + i * 256;
        int m = idx >> 3;              // 8 quads per 32-wide row
        int k = (idx & 7) * 4;
        const float* p = &A[(size_t)(m0 + m) * lda + (k0 + k)];
        r[i] = *(const float4*)p;
        if (k0 + BK < K) __builtin_prefetch(p + BK, 0, 1);   // global_prefetch_b8
    }
}

// B tile: BK x BN = 2048 floats = 512 float4; clamped, branch-free addressing.
// TRANSB: element (k,n) = B[n*ldb+k]  -> load quads along k
// !TRANSB: element (k,n) = B[k*ldb+n] -> load quads along n
template <bool TRANSB>
__device__ __forceinline__ void stage_load_B(const float* __restrict__ B, int ldb,
                                             int n0, int k0, int N, int tid, float4 r[2])
{
    #pragma unroll
    for (int i = 0; i < 2; ++i) {
        int idx = tid + i * 256;
        if (TRANSB) {
            int n = idx >> 3;          // 8 quads per 32-wide k-row
            int k = (idx & 7) * 4;
            int col = n0 + n; col = col < N ? col : N - 1;   // clamp (garbage cols masked later)
            r[i] = *(const float4*)&B[(size_t)col * ldb + (k0 + k)];
        } else {
            int k = idx >> 4;          // 16 quads per 64-wide n-row
            int n4 = (idx & 15) * 4;
            int col = n0 + n4; col = col <= N - 4 ? col : N - 4;  // N % 4 == 0 for all uses
            r[i] = *(const float4*)&B[(size_t)(k0 + k) * ldb + col];
        }
    }
}

__device__ __forceinline__ void stage_store_A(__bf16 (*As)[LDS_STRIDE], int tid,
                                              const float4 r[2])
{
    #pragma unroll
    for (int i = 0; i < 2; ++i) {
        int idx = tid + i * 256;
        int m = idx >> 3;
        int k = (idx & 7) * 4;
        As[m][k + 0] = (__bf16)r[i].x;
        As[m][k + 1] = (__bf16)r[i].y;
        As[m][k + 2] = (__bf16)r[i].z;
        As[m][k + 3] = (__bf16)r[i].w;
    }
}

template <bool TRANSB>
__device__ __forceinline__ void stage_store_B(__bf16 (*Bs)[LDS_STRIDE], int tid,
                                              const float4 r[2])
{
    #pragma unroll
    for (int i = 0; i < 2; ++i) {
        int idx = tid + i * 256;
        if (TRANSB) {
            int n = idx >> 3;
            int k = (idx & 7) * 4;
            Bs[n][k + 0] = (__bf16)r[i].x;
            Bs[n][k + 1] = (__bf16)r[i].y;
            Bs[n][k + 2] = (__bf16)r[i].z;
            Bs[n][k + 3] = (__bf16)r[i].w;
        } else {
            int k = idx >> 4;
            int n4 = (idx & 15) * 4;
            Bs[n4 + 0][k] = (__bf16)r[i].x;
            Bs[n4 + 1][k] = (__bf16)r[i].y;
            Bs[n4 + 2][k] = (__bf16)r[i].z;
            Bs[n4 + 3][k] = (__bf16)r[i].w;
        }
    }
}

// ---------------------------------------------------------------- WMMA GEMM
// C[M,N](f32, ldc) = A[M,K](f32, lda) * B (f32) (+bias) (+=C if accum)
// B element (k,n) = TRANSB ? B[n*ldb + k] : B[k*ldb + n]
// If cap_len != nullptr, rows with t >= cap_len[row]-1 are forced to 0.
// Requires: M % 64 == 0, K % 32 == 0, N % 4 == 0, lda/ldb % 4 == 0.
template <bool TRANSB>
__global__ __launch_bounds__(256)
void gemm_wmma_bf16(const float* __restrict__ A, int lda,
                    const float* __restrict__ B, int ldb,
                    float* __restrict__ C, int ldc,
                    const float* __restrict__ bias,
                    int M, int N, int K, int accum,
                    const int* __restrict__ cap_len, int t)
{
    __shared__ __bf16 As[2][BM][LDS_STRIDE];
    __shared__ __bf16 Bs[2][BN][LDS_STRIDE];   // stored transposed: [n][k]

    const int tid  = threadIdx.x;
    const int lane = tid & 31;
    const int wave = tid >> 5;              // 0..7
    const int wm   = wave >> 1;             // 0..3  (16-row tile)
    const int wn   = wave & 1;              // 0..1  (32-col tile)
    const int half = lane >> 4;             // 0/1
    const int l16  = lane & 15;

    const int m0 = blockIdx.y * BM;
    const int n0 = blockIdx.x * BN;

    const int am  = wm * 16 + l16;          // A fragment row within block tile
    const int bn0 = wn * 32 + l16;          // B fragment col 0 within block tile
    const int bn1 = bn0 + 16;               // B fragment col 1

    v8f acc0 = {}, acc1 = {};
    float4 aR[2], bR[2];

    // prologue: stage tile 0
    stage_load_A(A, lda, m0, 0, K, tid, aR);
    stage_load_B<TRANSB>(B, ldb, n0, 0, N, tid, bR);
    stage_store_A(As[0], tid, aR);
    stage_store_B<TRANSB>(Bs[0], tid, bR);

    const int nk = K / BK;

    // steady state: unconditional next-tile staging (last iteration peeled)
    for (int kt = 0; kt < nk - 1; ++kt) {
        __syncthreads();                    // LDS[cur] ready for all waves
        const int cur = kt & 1, nxt = cur ^ 1;

        // issue next tile's global loads before the matrix math (latency overlap)
        stage_load_A(A, lda, m0, (kt + 1) * BK, K, tid, aR);
        stage_load_B<TRANSB>(B, ldb, n0, (kt + 1) * BK, N, tid, bR);

        // build fragments (ISA 7.12.2 layouts) and issue WMMAs
        v16bf af, bf0, bf1;
        #pragma unroll
        for (int v = 0; v < 8; ++v) {
            int ka = 16 * (v >> 2) + 8 * half + 2 * (v & 3);   // 16-bit A 16x32 layout
            af[2 * v]     = As[cur][am][ka];
            af[2 * v + 1] = As[cur][am][ka + 1];
            int kb = 16 * (v >> 2) + 8 * half + 2 * (v & 3);   // 16-bit B 32x16 layout (same k order as A)
            bf0[2 * v]     = Bs[cur][bn0][kb];
            bf0[2 * v + 1] = Bs[cur][bn0][kb + 1];
            bf1[2 * v]     = Bs[cur][bn1][kb];
            bf1[2 * v + 1] = Bs[cur][bn1][kb + 1];
        }
        acc0 = __builtin_amdgcn_wmma_f32_16x16x32_bf16(
                   false, af, false, bf0, (short)0, acc0, false, false);
        acc1 = __builtin_amdgcn_wmma_f32_16x16x32_bf16(
                   false, af, false, bf1, (short)0, acc1, false, false);

        // convert + store next tile into the other LDS buffer
        stage_store_A(As[nxt], tid, aR);
        stage_store_B<TRANSB>(Bs[nxt], tid, bR);
    }

    // final K tile (no staging)
    {
        __syncthreads();
        const int cur = (nk - 1) & 1;
        v16bf af, bf0, bf1;
        #pragma unroll
        for (int v = 0; v < 8; ++v) {
            int ka = 16 * (v >> 2) + 8 * half + 2 * (v & 3);
            af[2 * v]     = As[cur][am][ka];
            af[2 * v + 1] = As[cur][am][ka + 1];
            int kb = 16 * (v >> 2) + 8 * half + 2 * (v & 3);
            bf0[2 * v]     = Bs[cur][bn0][kb];
            bf0[2 * v + 1] = Bs[cur][bn0][kb + 1];
            bf1[2 * v]     = Bs[cur][bn1][kb];
            bf1[2 * v + 1] = Bs[cur][bn1][kb + 1];
        }
        acc0 = __builtin_amdgcn_wmma_f32_16x16x32_bf16(
                   false, af, false, bf0, (short)0, acc0, false, false);
        acc1 = __builtin_amdgcn_wmma_f32_16x16x32_bf16(
                   false, af, false, bf1, (short)0, acc1, false, false);
    }

    // ---- epilogue: C/D layout: VGPR r -> row half*8+r, col = lane&15
    #pragma unroll
    for (int sub = 0; sub < 2; ++sub) {
        const int col = n0 + wn * 32 + sub * 16 + l16;
        if (col < N) {
            const v8f acc = sub ? acc1 : acc0;
            #pragma unroll
            for (int r = 0; r < 8; ++r) {
                int row = m0 + wm * 16 + half * 8 + r;
                float v = acc[r];
                if (bias)  v += bias[col];
                if (accum) v += C[(size_t)row * ldc + col];
                if (cap_len && t >= cap_len[row] - 1) v = 0.0f;    // inactive row mask
                C[(size_t)row * ldc + col] = v;
            }
        }
    }
}

// ---------------------------------------------------------------- small kernels
__global__ void mean_enc_kernel(const float* __restrict__ enc, float* __restrict__ mean)
{
    int gid = blockIdx.x * blockDim.x + threadIdx.x;   // 64*2048
    int b = gid >> 11, e = gid & 2047;
    const float* p = enc + ((size_t)b * PIX) * ENC + e;
    float s = 0.0f;
    for (int i = 0; i < PIX; ++i) s += p[(size_t)i * ENC];
    mean[gid] = s * (1.0f / (float)PIX);
}

// one wave per (b,p): e[b,p] = sum_a relu(att1+att2)*v_att + bv
__global__ __launch_bounds__(256)
void attn_scores_kernel(const float* __restrict__ att1, const float* __restrict__ att2,
                        const float* __restrict__ v_att, const float* __restrict__ bv,
                        float* __restrict__ e)
{
    int gw   = blockIdx.x * (blockDim.x >> 5) + (threadIdx.x >> 5);
    int lane = threadIdx.x & 31;
    if (gw >= BATCH * PIX) return;
    int b = gw / PIX;
    const float* a1 = att1 + (size_t)gw * ATT;
    const float* a2 = att2 + (size_t)b * ATT;
    float s = 0.0f;
    #pragma unroll 4
    for (int j = lane; j < ATT; j += 32) {
        float v = a1[j] + a2[j];
        v = v > 0.0f ? v : 0.0f;
        s += v * v_att[j];
    }
    #pragma unroll
    for (int off = 16; off; off >>= 1) s += __shfl_xor(s, off, 32);
    if (lane == 0) e[gw] = s + bv[0];
}

// one block per batch row: softmax over 196 pixels, write alpha ws + masked output
__global__ __launch_bounds__(256)
void attn_softmax_kernel(const float* __restrict__ e, float* __restrict__ alpha,
                         float* __restrict__ alpha_out,
                         const int* __restrict__ cap_len, int t)
{
    __shared__ float red[8];
    __shared__ float s_max, s_sum;
    int b = blockIdx.x, tid = threadIdx.x, lane = tid & 31, wv = tid >> 5;

    float v = (tid < PIX) ? e[b * PIX + tid] : -3.4e38f;
    float m = v;
    #pragma unroll
    for (int off = 16; off; off >>= 1) m = fmaxf(m, __shfl_xor(m, off, 32));
    if (lane == 0) red[wv] = m;
    __syncthreads();
    if (tid == 0) {
        float mm = red[0];
        for (int i = 1; i < 8; ++i) mm = fmaxf(mm, red[i]);
        s_max = mm;
    }
    __syncthreads();

    float ex = (tid < PIX) ? __expf(v - s_max) : 0.0f;
    float s = ex;
    #pragma unroll
    for (int off = 16; off; off >>= 1) s += __shfl_xor(s, off, 32);
    if (lane == 0) red[wv] = s;
    __syncthreads();
    if (tid == 0) {
        float ss = 0.0f;
        for (int i = 0; i < 8; ++i) ss += red[i];
        s_sum = ss;
    }
    __syncthreads();

    if (tid < PIX) {
        float a = ex / s_sum;
        alpha[b * PIX + tid] = a;
        bool active = t < cap_len[b] - 1;
        alpha_out[(size_t)b * TSTEPS * PIX + (size_t)t * PIX + tid] = active ? a : 0.0f;
    }
}

// ctx[b,e] = sigmoid(fbeta_raw[b,e]) * sum_p alpha[b,p]*enc[b,p,e] -> xcat[:,512+e]
__global__ void ctx_gate_kernel(const float* __restrict__ alpha, const float* __restrict__ enc,
                                const float* __restrict__ fbeta_raw, float* __restrict__ xcat)
{
    int gid = blockIdx.x * blockDim.x + threadIdx.x;   // 64*2048
    int b = gid >> 11, e = gid & 2047;
    const float* al = alpha + b * PIX;
    const float* ep = enc + ((size_t)b * PIX) * ENC + e;
    float s = 0.0f;
    #pragma unroll 4
    for (int p = 0; p < PIX; ++p) s += al[p] * ep[(size_t)p * ENC];
    xcat[(size_t)b * XDIM + 512 + e] = sigmoidf_(fbeta_raw[gid]) * s;
}

__global__ void embed_kernel(const int* __restrict__ captions, const float* __restrict__ emb,
                             float* __restrict__ xcat, int t)
{
    int gid = blockIdx.x * blockDim.x + threadIdx.x;   // 64*512
    int b = gid >> 9, j = gid & 511;
    int tok = captions[b * TCAP + t];
    xcat[(size_t)b * XDIM + j] = emb[(size_t)tok * DEC + j];
}

__global__ void lstm_pointwise_kernel(const float* __restrict__ g,
                                      float* __restrict__ h, float* __restrict__ c,
                                      const int* __restrict__ cap_len, int t)
{
    int gid = blockIdx.x * blockDim.x + threadIdx.x;   // 64*512
    int b = gid >> 9, d = gid & 511;
    const float* gr = g + (size_t)b * (4 * DEC);
    float ig = sigmoidf_(gr[d]);
    float fg = sigmoidf_(gr[DEC + d]);
    float gg = tanhf(gr[2 * DEC + d]);
    float og = sigmoidf_(gr[3 * DEC + d]);
    float cn = fg * c[gid] + ig * gg;
    float hn = og * tanhf(cn);
    if (t < cap_len[b] - 1) { c[gid] = cn; h[gid] = hn; }   // carry old state when inactive
}

__global__ void tail_copy_kernel(const int* __restrict__ captions,
                                 const int* __restrict__ cap_len,
                                 float* __restrict__ out_caps, float* __restrict__ out_lens)
{
    int gid = blockIdx.x * blockDim.x + threadIdx.x;
    if (gid < BATCH * TCAP) out_caps[gid] = (float)captions[gid];
    if (gid < BATCH)        out_lens[gid] = (float)(cap_len[gid] - 1);
}

// ---------------------------------------------------------------- host driver
extern "C" void kernel_launch(void* const* d_in, const int* in_sizes, int n_in,
                              void* d_out, int out_size, void* d_ws, size_t ws_size,
                              hipStream_t stream)
{
    const float* enc      = (const float*)d_in[0];
    const int*   captions = (const int*)  d_in[1];
    const int*   cap_len  = (const int*)  d_in[2];
    const float* emb      = (const float*)d_in[3];
    const float* We_att   = (const float*)d_in[4];
    const float* be_att   = (const float*)d_in[5];
    const float* Wd_att   = (const float*)d_in[6];
    const float* bd_att   = (const float*)d_in[7];
    const float* v_att    = (const float*)d_in[8];
    const float* bv_att   = (const float*)d_in[9];
    const float* W_ih     = (const float*)d_in[10];
    const float* b_ih     = (const float*)d_in[11];
    const float* W_hh     = (const float*)d_in[12];
    const float* b_hh     = (const float*)d_in[13];
    const float* W_init_h = (const float*)d_in[14];
    const float* b_init_h = (const float*)d_in[15];
    const float* W_init_c = (const float*)d_in[16];
    const float* b_init_c = (const float*)d_in[17];
    const float* W_fbeta  = (const float*)d_in[18];
    const float* b_fbeta  = (const float*)d_in[19];
    const float* W_fc     = (const float*)d_in[20];
    const float* b_fc     = (const float*)d_in[21];

    float* out = (float*)d_out;
    // output layout: preds(64*24*10000) | captions(1600) | lens(64) | alphas(64*24*196)
    float* out_caps   = out + (size_t)BATCH * TSTEPS * VOCAB;
    float* out_lens   = out_caps + BATCH * TCAP;
    float* out_alphas = out_lens + BATCH;

    // workspace carve-up (floats)
    float* ws = (float*)d_ws;
    float* w_mean  = ws;                         // 64*2048
    float* w_h     = w_mean  + BATCH * ENC;      // 64*512
    float* w_c     = w_h     + BATCH * DEC;      // 64*512
    float* w_att2  = w_c     + BATCH * DEC;      // 64*512
    float* w_e     = w_att2  + BATCH * ATT;      // 64*196
    float* w_alpha = w_e     + BATCH * PIX;      // 64*196
    float* w_fbeta = w_alpha + BATCH * PIX;      // 64*2048
    float* w_xcat  = w_fbeta + BATCH * ENC;      // 64*2560
    float* w_g     = w_xcat  + (size_t)BATCH * XDIM;   // 64*2048
    float* w_att1  = w_g     + BATCH * 4 * DEC;  // 12544*512

    #define GEMM_N(A, lda, Bm, ldb, Cm, ldc, bias, M, N, K, accum, mask, tt)          \
        gemm_wmma_bf16<false><<<dim3(((N) + BN - 1) / BN, (M) / BM), 256, 0, stream>>>( \
            A, lda, Bm, ldb, Cm, ldc, bias, M, N, K, accum, mask, tt)
    #define GEMM_T(A, lda, Bm, ldb, Cm, ldc, bias, M, N, K, accum, mask, tt)          \
        gemm_wmma_bf16<true><<<dim3(((N) + BN - 1) / BN, (M) / BM), 256, 0, stream>>>( \
            A, lda, Bm, ldb, Cm, ldc, bias, M, N, K, accum, mask, tt)

    // ---- precompute
    mean_enc_kernel<<<(BATCH * ENC) / 256, 256, 0, stream>>>(enc, w_mean);
    GEMM_N(w_mean, ENC, W_init_h, DEC, w_h, DEC, b_init_h, BATCH, DEC, ENC, 0, (const int*)nullptr, 0);
    GEMM_N(w_mean, ENC, W_init_c, DEC, w_c, DEC, b_init_c, BATCH, DEC, ENC, 0, (const int*)nullptr, 0);
    GEMM_N(enc, ENC, We_att, ATT, w_att1, ATT, be_att, BATCH * PIX, ATT, ENC, 0, (const int*)nullptr, 0);

    // ---- recurrence
    for (int t = 0; t < TSTEPS; ++t) {
        embed_kernel<<<(BATCH * DEC) / 256, 256, 0, stream>>>(captions, emb, w_xcat, t);
        GEMM_N(w_h, DEC, Wd_att, ATT, w_att2, ATT, bd_att, BATCH, ATT, DEC, 0, (const int*)nullptr, 0);
        attn_scores_kernel<<<(BATCH * PIX + 7) / 8, 256, 0, stream>>>(w_att1, w_att2, v_att, bv_att, w_e);
        attn_softmax_kernel<<<BATCH, 256, 0, stream>>>(w_e, w_alpha, out_alphas, cap_len, t);
        GEMM_N(w_h, DEC, W_fbeta, ENC, w_fbeta, ENC, b_fbeta, BATCH, ENC, DEC, 0, (const int*)nullptr, 0);
        ctx_gate_kernel<<<(BATCH * ENC) / 256, 256, 0, stream>>>(w_alpha, enc, w_fbeta, w_xcat);
        GEMM_T(w_xcat, XDIM, W_ih, XDIM, w_g, 4 * DEC, b_ih, BATCH, 4 * DEC, XDIM, 0, (const int*)nullptr, 0);
        GEMM_T(w_h, DEC, W_hh, DEC, w_g, 4 * DEC, b_hh, BATCH, 4 * DEC, DEC, 1, (const int*)nullptr, 0);
        lstm_pointwise_kernel<<<(BATCH * DEC) / 256, 256, 0, stream>>>(w_g, w_h, w_c, cap_len, t);
        GEMM_N(w_h, DEC, W_fc, VOCAB, out + (size_t)t * VOCAB, TSTEPS * VOCAB, b_fc,
               BATCH, VOCAB, DEC, 0, cap_len, t);
    }

    tail_copy_kernel<<<(BATCH * TCAP + 255) / 256, 256, 0, stream>>>(captions, cap_len, out_caps, out_lens);
    #undef GEMM_N
    #undef GEMM_T
}
